// LLaDAAttention_11665131176295
// MI455X (gfx1250) — matrix operand order repaired and checked by reading, not compiled, not run on hardware
//
#include <hip/hip_runtime.h>
#include <stdint.h>

typedef __bf16 bf16_t;
typedef __attribute__((ext_vector_type(2)))  __bf16 bf16x2;
typedef __attribute__((ext_vector_type(4)))  __bf16 bf16x4;
typedef __attribute__((ext_vector_type(8)))  __bf16 bf16x8;
typedef __attribute__((ext_vector_type(16))) __bf16 bf16x16;
typedef __attribute__((ext_vector_type(4)))  float  floatx4;
typedef __attribute__((ext_vector_type(8)))  float  floatx8;

#define B_DIM   2
#define S_DIM   2048
#define E_DIM   2048
#define H_DIM   16
#define D_HEAD  128
#define N_QKV   (3 * E_DIM)          // 6144
#define ROWS    (B_DIM * S_DIM)      // 4096
#define QKV_ELEMS ((size_t)B_DIM * H_DIM * S_DIM * D_HEAD)   // 8,388,608 bf16 each

static __device__ __forceinline__ floatx8 fzero() {
  floatx8 z;
#pragma unroll
  for (int i = 0; i < 8; ++i) z[i] = 0.0f;
  return z;
}

static __device__ __forceinline__ bf16x16 join16(bf16x8 lo, bf16x8 hi) {
  return __builtin_shufflevector(lo, hi, 0,1,2,3,4,5,6,7,8,9,10,11,12,13,14,15);
}

// A fragment: 16x32 bf16, row-major source [16][K], lda in elements.
// ISA layout: lane = m + 16h; VGPR0-3 hold K = 8h..8h+7, VGPR4-7 hold K = 16+8h..16+8h+7.
static __device__ __forceinline__ bf16x16 load_a_frag(const bf16_t* a, int lda) {
  const int lane = threadIdx.x & 31;
  const int m = lane & 15, h = lane >> 4;
  const bf16_t* r = a + m * lda + h * 8;
  bf16x8 lo = *(const bf16x8*)(r);
  bf16x8 hi = *(const bf16x8*)(r + 16);
  return join16(lo, hi);
}

// B fragment: 32x16 bf16 from transposed staging Bt[n][k], ldb in elements.
// ISA layout: lane = n + 16h; VGPR0-3 hold K = 8h..8h+7, VGPR4-7 hold K = 16+8h..16+8h+7.
static __device__ __forceinline__ bf16x16 load_b_frag(const bf16_t* bt, int ldb) {
  const int lane = threadIdx.x & 31;
  const int n = lane & 15, h = lane >> 4;
  const bf16_t* r = bt + n * ldb + h * 8;
  bf16x8 lo = *(const bf16x8*)(r);
  bf16x8 hi = *(const bf16x8*)(r + 16);
  return join16(lo, hi);
}

static __device__ __forceinline__ floatx8 wmma_bf16(bf16x16 a, bf16x16 b, floatx8 c) {
  return __builtin_amdgcn_wmma_f32_16x16x32_bf16(false, a, false, b, (short)0, c, false, false);
}

// gfx1250 async global->LDS copy (16B per lane), tracked on ASYNCcnt.
// VDST = LDS byte address (low 32 bits of the generic shared pointer per the
// flat-aperture truncation rule), VADDR = 64-bit global address.
static __device__ __forceinline__ void async_copy_b128(bf16_t* lds_dst, const bf16_t* g_src) {
  uint32_t lds_off = (uint32_t)(uintptr_t)lds_dst;
  asm volatile("global_load_async_to_lds_b128 %0, %1, off"
               :: "v"(lds_off), "v"(g_src) : "memory");
}
static __device__ __forceinline__ void wait_async0() {
  asm volatile("s_wait_asynccnt 0x0" ::: "memory");
}

// ---------------------------------------------------------------------------
// Kernel 1: qkv = x @ w_qkv, epilogue scatters bf16 into q/k/v [B,H,S,D]
// ---------------------------------------------------------------------------
#define BM 128
#define BN 128
#define BK 32
#define LDA 40   // 32 + 8 pad, 80B rows (16B aligned)
#define LDB 40

__global__ __launch_bounds__(256) void gemm_qkv_kernel(
    const float* __restrict__ X, const float* __restrict__ W,
    bf16_t* __restrict__ qb, bf16_t* __restrict__ kb, bf16_t* __restrict__ vb) {
  __shared__ __align__(16) bf16_t As[BM * LDA];
  __shared__ __align__(16) bf16_t Bs[BN * LDB];
  const int n0 = blockIdx.x * BN;
  const int m0 = blockIdx.y * BM;
  const int tid = threadIdx.x;
  const int wave = tid >> 5;
  const int lane = tid & 31;
  const int wm = (wave & 3) * 32;   // wave row offset (4 waves in M)
  const int wn = (wave >> 2) * 64;  // wave col offset (2 waves in N)

  floatx8 acc[2][4];
#pragma unroll
  for (int i = 0; i < 2; ++i)
#pragma unroll
    for (int j = 0; j < 4; ++j) acc[i][j] = fzero();

  for (int k0 = 0; k0 < E_DIM; k0 += BK) {
    // A tile: float4 loads -> packed bf16x4 LDS stores (4 vec-ops/thread)
#pragma unroll
    for (int it = 0; it < (BM * BK) / (256 * 4); ++it) {
      const int vi = tid + it * 256;
      const int r = vi >> 3, c4 = vi & 7;
      floatx4 xv = *(const floatx4*)&X[(size_t)(m0 + r) * E_DIM + k0 + c4 * 4];
      bf16x4 bv;
#pragma unroll
      for (int e = 0; e < 4; ++e) bv[e] = (bf16_t)xv[e];
      *(bf16x4*)&As[r * LDA + c4 * 4] = bv;
    }
    // B tile transposed: two coalesced b32 loads -> bf16x2 store (8/thread)
#pragma unroll
    for (int it = 0; it < (BN * (BK / 2)) / 256; ++it) {
      const int vi = tid + it * 256;
      const int nn = vi & 127, kp = vi >> 7;
      const float w0 = W[(size_t)(k0 + 2 * kp) * N_QKV + n0 + nn];
      const float w1 = W[(size_t)(k0 + 2 * kp + 1) * N_QKV + n0 + nn];
      bf16x2 bv;
      bv[0] = (bf16_t)w0;
      bv[1] = (bf16_t)w1;
      *(bf16x2*)&Bs[nn * LDB + 2 * kp] = bv;
    }
    __syncthreads();

    bf16x16 af[2], bfr[4];
#pragma unroll
    for (int i = 0; i < 2; ++i) af[i] = load_a_frag(&As[(wm + i * 16) * LDA], LDA);
#pragma unroll
    for (int j = 0; j < 4; ++j) bfr[j] = load_b_frag(&Bs[(wn + j * 16) * LDB], LDB);
#pragma unroll
    for (int i = 0; i < 2; ++i)
#pragma unroll
      for (int j = 0; j < 4; ++j) acc[i][j] = wmma_bf16(af[i], bfr[j], acc[i][j]);
    __syncthreads();
  }

  const int hl = lane >> 4, nl = lane & 15;
#pragma unroll
  for (int i = 0; i < 2; ++i)
#pragma unroll
    for (int j = 0; j < 4; ++j)
#pragma unroll
      for (int r = 0; r < 8; ++r) {
        const int row = m0 + wm + i * 16 + r + 8 * hl;
        const int col = n0 + wn + j * 16 + nl;
        const int b_ = row >> 11, s = row & 2047;
        const int which = col >> 11, e = col & 2047;
        const int h = e >> 7, d = e & 127;
        bf16_t* dst = (which == 0) ? qb : ((which == 1) ? kb : vb);
        dst[(((size_t)b_ * H_DIM + h) * S_DIM + s) * D_HEAD + d] = (bf16_t)acc[i][j][r];
      }
}

// ---------------------------------------------------------------------------
// Kernel 2: in-place RoPE on q and k, [B,H,S,D] bf16
// ---------------------------------------------------------------------------
__global__ __launch_bounds__(256) void rope_kernel(bf16_t* __restrict__ q,
                                                   bf16_t* __restrict__ k) {
  const int idx = blockIdx.x * 256 + threadIdx.x;   // B*H*S*64 threads
  const int d  = idx & 63;
  const int s  = (idx >> 6) & 2047;
  const int bh = idx >> 17;
  const size_t base = ((size_t)bh * S_DIM + s) * D_HEAD;
  // inv_freq = 10000^(-d/64) = exp(-d * ln(10000)/64)
  const float freq = __expf(-(float)d * (9.210340371976184f / 64.0f));
  const float ang = (float)s * freq;
  float sn, cs;
  __sincosf(ang, &sn, &cs);

  float q1 = (float)q[base + d], q2 = (float)q[base + d + 64];
  q[base + d]      = (bf16_t)(q1 * cs - q2 * sn);
  q[base + d + 64] = (bf16_t)(q2 * cs + q1 * sn);

  float k1 = (float)k[base + d], k2 = (float)k[base + d + 64];
  k[base + d]      = (bf16_t)(k1 * cs - k2 * sn);
  k[base + d + 64] = (bf16_t)(k2 * cs + k1 * sn);
}

// ---------------------------------------------------------------------------
// Kernel 3: flash attention per (b,h,q-tile). 128 threads = 4 waves,
// each wave owns 16 q rows. KV tiles of 64. Writes y [B,S,E] bf16.
// Q and K tiles stage via async global->LDS b128 copies (ASYNCcnt).
// ---------------------------------------------------------------------------
#define QT 64
#define KT 64
#define LQ (D_HEAD + 8)   // 136 elems -> 272B rows (16B multiple)
#define LK (D_HEAD + 8)
#define LV (KT + 8)       // 72 elems -> 144B rows (16B multiple)
#define LP (KT + 8)

__global__ __launch_bounds__(128) void attn_kernel(
    const bf16_t* __restrict__ Q, const bf16_t* __restrict__ K,
    const bf16_t* __restrict__ V, bf16_t* __restrict__ Yo) {
  __shared__ __align__(16) bf16_t Qs[QT * LQ];
  __shared__ __align__(16) bf16_t Ks[KT * LK];
  __shared__ __align__(16) bf16_t Vs[D_HEAD * LV];   // transposed: [d][kv]
  __shared__ __align__(16) bf16_t Ps[QT * LP];       // per-wave 16-row slabs

  const int qt = blockIdx.x, bh = blockIdx.y;
  const size_t base = (size_t)bh * S_DIM * D_HEAD;
  const int tid = threadIdx.x, wave = tid >> 5, lane = tid & 31;
  const int hl = lane >> 4, nl = lane & 15;
  const int q0 = qt * QT;
  const float scale = 0.08838834764831845f;   // 1/sqrt(128)

  // Q tile: raw async byte copy (scale folded into post-WMMA pass)
#pragma unroll
  for (int it = 0; it < (QT * D_HEAD) / (128 * 8); ++it) {   // 8 copies/thread
    const int idx = tid + it * 128;
    const int r = idx >> 4, c8 = idx & 15;
    async_copy_b128(&Qs[r * LQ + c8 * 8],
                    &Q[base + (size_t)(q0 + r) * D_HEAD + c8 * 8]);
  }

  floatx8 o[8];
#pragma unroll
  for (int i = 0; i < 8; ++i) o[i] = fzero();
  float mst[8], lst[8];
#pragma unroll
  for (int r = 0; r < 8; ++r) { mst[r] = -1e30f; lst[r] = 0.0f; }

  for (int kv0 = 0; kv0 < S_DIM; kv0 += KT) {
    __syncthreads();   // previous tile fully consumed
    // K tile: async row copies
#pragma unroll
    for (int it = 0; it < (KT * D_HEAD) / (128 * 8); ++it) {
      const int idx = tid + it * 128;
      const int r = idx >> 4, c8 = idx & 15;
      async_copy_b128(&Ks[r * LK + c8 * 8],
                      &K[base + (size_t)(kv0 + r) * D_HEAD + c8 * 8]);
    }
    // V tile: vector load + transposed scalar LDS stores
#pragma unroll
    for (int it = 0; it < (KT * D_HEAD) / (128 * 8); ++it) {
      const int idx = tid + it * 128;
      const int r = idx >> 4, c8 = idx & 15;
      bf16x8 vv = *(const bf16x8*)&V[base + (size_t)(kv0 + r) * D_HEAD + c8 * 8];
#pragma unroll
      for (int e = 0; e < 8; ++e) Vs[(c8 * 8 + e) * LV + r] = vv[e];
    }
    wait_async0();
    __syncthreads();

    // scores: Q(16x128) @ K^T(128x64) per wave
    floatx8 sc[4];
#pragma unroll
    for (int j = 0; j < 4; ++j) sc[j] = fzero();
    for (int ks = 0; ks < D_HEAD; ks += 32) {
      bf16x16 af = load_a_frag(&Qs[(wave * 16) * LQ + ks], LQ);
#pragma unroll
      for (int j = 0; j < 4; ++j) {
        bf16x16 bfr = load_b_frag(&Ks[(j * 16) * LK + ks], LK);
        sc[j] = wmma_bf16(af, bfr, sc[j]);
      }
    }
#pragma unroll
    for (int j = 0; j < 4; ++j)
#pragma unroll
      for (int r = 0; r < 8; ++r) sc[j][r] *= scale;

    // online softmax; row M = r + 8*hl, reductions over the 16 lanes of same hl
    float mnew[8], alpha[8];
#pragma unroll
    for (int r = 0; r < 8; ++r) {
      float v_ = fmaxf(fmaxf(sc[0][r], sc[1][r]), fmaxf(sc[2][r], sc[3][r]));
      for (int off = 1; off < 16; off <<= 1) v_ = fmaxf(v_, __shfl_xor(v_, off, 16));
      mnew[r] = fmaxf(mst[r], v_);
      alpha[r] = __expf(mst[r] - mnew[r]);
      mst[r] = mnew[r];
    }
    float psum[8];
#pragma unroll
    for (int r = 0; r < 8; ++r) psum[r] = 0.0f;
#pragma unroll
    for (int j = 0; j < 4; ++j)
#pragma unroll
      for (int r = 0; r < 8; ++r) {
        const float p = __expf(sc[j][r] - mnew[r]);
        psum[r] += p;
        Ps[(wave * 16 + r + 8 * hl) * LP + j * 16 + nl] = (bf16_t)p;
      }
#pragma unroll
    for (int r = 0; r < 8; ++r) {
      float ps = psum[r];
      for (int off = 1; off < 16; off <<= 1) ps += __shfl_xor(ps, off, 16);
      lst[r] = lst[r] * alpha[r] + ps;
#pragma unroll
      for (int of = 0; of < 8; ++of) o[of][r] *= alpha[r];
    }

    // O += P(16x64) @ V(64x128)
#pragma unroll
    for (int of = 0; of < 8; ++of) {
      for (int ks = 0; ks < KT; ks += 32) {
        bf16x16 af = load_a_frag(&Ps[(wave * 16) * LP + ks], LP);
        bf16x16 bfr = load_b_frag(&Vs[(of * 16) * LV + ks], LV);
        o[of] = wmma_bf16(af, bfr, o[of]);
      }
    }
  }

  const int b_ = bh >> 4, h = bh & 15;
#pragma unroll
  for (int of = 0; of < 8; ++of)
#pragma unroll
    for (int r = 0; r < 8; ++r) {
      const int srow = q0 + wave * 16 + r + 8 * hl;
      const int d = of * 16 + nl;
      const float val = o[of][r] / lst[r];
      Yo[((size_t)(b_ * S_DIM + srow)) * E_DIM + h * D_HEAD + d] = (bf16_t)val;
    }
}

// ---------------------------------------------------------------------------
// Kernel 4: out = y @ w_proj  (bf16 A from ws, f32 W converted, f32 out)
// ---------------------------------------------------------------------------
__global__ __launch_bounds__(256) void gemm_proj_kernel(
    const bf16_t* __restrict__ Y, const float* __restrict__ W,
    float* __restrict__ out) {
  __shared__ __align__(16) bf16_t As[BM * LDA];
  __shared__ __align__(16) bf16_t Bs[BN * LDB];
  const int n0 = blockIdx.x * BN;
  const int m0 = blockIdx.y * BM;
  const int tid = threadIdx.x;
  const int wave = tid >> 5;
  const int lane = tid & 31;
  const int wm = (wave & 3) * 32;
  const int wn = (wave >> 2) * 64;

  floatx8 acc[2][4];
#pragma unroll
  for (int i = 0; i < 2; ++i)
#pragma unroll
    for (int j = 0; j < 4; ++j) acc[i][j] = fzero();

  for (int k0 = 0; k0 < E_DIM; k0 += BK) {
    // A tile (already bf16): b128 copies, 2 per thread
#pragma unroll
    for (int it = 0; it < (BM * BK) / (256 * 8); ++it) {
      const int vi = tid + it * 256;
      const int r = vi >> 2, c8 = vi & 3;
      bf16x8 av = *(const bf16x8*)&Y[(size_t)(m0 + r) * E_DIM + k0 + c8 * 8];
      *(bf16x8*)&As[r * LDA + c8 * 8] = av;
    }
    // B tile transposed, pair-packed
#pragma unroll
    for (int it = 0; it < (BN * (BK / 2)) / 256; ++it) {
      const int vi = tid + it * 256;
      const int nn = vi & 127, kp = vi >> 7;
      const float w0 = W[(size_t)(k0 + 2 * kp) * E_DIM + n0 + nn];
      const float w1 = W[(size_t)(k0 + 2 * kp + 1) * E_DIM + n0 + nn];
      bf16x2 bv;
      bv[0] = (bf16_t)w0;
      bv[1] = (bf16_t)w1;
      *(bf16x2*)&Bs[nn * LDB + 2 * kp] = bv;
    }
    __syncthreads();

    bf16x16 af[2], bfr[4];
#pragma unroll
    for (int i = 0; i < 2; ++i) af[i] = load_a_frag(&As[(wm + i * 16) * LDA], LDA);
#pragma unroll
    for (int j = 0; j < 4; ++j) bfr[j] = load_b_frag(&Bs[(wn + j * 16) * LDB], LDB);
#pragma unroll
    for (int i = 0; i < 2; ++i)
#pragma unroll
      for (int j = 0; j < 4; ++j) acc[i][j] = wmma_bf16(af[i], bfr[j], acc[i][j]);
    __syncthreads();
  }

  const int hl = lane >> 4, nl = lane & 15;
#pragma unroll
  for (int i = 0; i < 2; ++i)
#pragma unroll
    for (int j = 0; j < 4; ++j)
#pragma unroll
      for (int r = 0; r < 8; ++r) {
        const int row = m0 + wm + i * 16 + r + 8 * hl;
        const int col = n0 + wn + j * 16 + nl;
        out[(size_t)row * E_DIM + col] = acc[i][j][r];
      }
}

// ---------------------------------------------------------------------------
extern "C" void kernel_launch(void* const* d_in, const int* in_sizes, int n_in,
                              void* d_out, int out_size, void* d_ws, size_t ws_size,
                              hipStream_t stream) {
  (void)in_sizes; (void)n_in; (void)out_size; (void)ws_size;
  const float* x      = (const float*)d_in[0];
  const float* w_qkv  = (const float*)d_in[1];
  const float* w_proj = (const float*)d_in[2];
  float* out = (float*)d_out;

  char* ws = (char*)d_ws;
  const size_t qkv_bytes = QKV_ELEMS * sizeof(bf16_t);   // 16 MiB each
  bf16_t* qb = (bf16_t*)(ws);
  bf16_t* kb = (bf16_t*)(ws + qkv_bytes);
  bf16_t* vb = (bf16_t*)(ws + 2 * qkv_bytes);
  bf16_t* yb = (bf16_t*)(ws + 3 * qkv_bytes);

  // 1) qkv = x @ w_qkv  -> scatter q/k/v bf16 [B,H,S,D]
  gemm_qkv_kernel<<<dim3(N_QKV / BN, ROWS / BM), 256, 0, stream>>>(x, w_qkv, qb, kb, vb);
  // 2) RoPE in place on q,k
  rope_kernel<<<(B_DIM * H_DIM * S_DIM * (D_HEAD / 2)) / 256, 256, 0, stream>>>(qb, kb);
  // 3) flash attention -> y bf16 [B,S,E]
  attn_kernel<<<dim3(S_DIM / QT, B_DIM * H_DIM), 128, 0, stream>>>(qb, kb, vb, yb);
  // 4) out = y @ w_proj (f32 out)
  gemm_proj_kernel<<<dim3(E_DIM / BN, ROWS / BM), 256, 0, stream>>>(yb, w_proj, out);
}
